// ConvMDAformerBlock3D_63883343561398
// MI455X (gfx1250) — compile-verified, matrix operand order checked by reading, not run on hardware
//
#include <hip/hip_runtime.h>

// ---------------------------------------------------------------------------
// ConvMDAformer block for MI455X (gfx1250, wave32).
// All matmul-shaped stages (qkv, proj, grouped 3x3x3 conv as implicit GEMM,
// 1x1 conv, fc1, fc2) run on v_wmma_f32_16x16x32_bf16 with fp32 accumulate.
// Conv zero-padding is done by address selection against a zero page so the
// inner loop has no EXEC-divergent loads.  Workspace requirement: ~309 MB.
// ---------------------------------------------------------------------------

typedef __bf16 bf16_t;
typedef __attribute__((ext_vector_type(16))) __bf16 bf16x16;
typedef __attribute__((ext_vector_type(8)))  float   f32x8;

#define M_TOK  200704          // 4*16*56*56 tokens
#define NTOK_B 50176           // 16*56*56 tokens per batch image

union BPack { unsigned u; bf16_t h[2]; };

struct Frag { union { bf16x16 v; unsigned u[8]; }; };

// Load a 16x32 bf16 fragment from a row-major [rows][ld] matrix.
// A-matrix layout (ISA 7.12.2): lane L holds row (L&15); VGPR v holds the
// K pair  k = 2v + (v>=4 ? 8 : 0) + 8*(L>=16).  Pairs are contiguous in K,
// so v=0..3 and v=4..7 each coalesce into one global_load_b128.
__device__ __forceinline__ void load_frag(Frag& f, const bf16_t* __restrict__ base,
                                          int row0, int ld, int k0, int lane) {
  const bf16_t* p = base + (size_t)(row0 + (lane & 15)) * ld
                         + (k0 + ((lane >> 4) << 3));
#pragma unroll
  for (int v = 0; v < 8; ++v)
    f.u[v] = *(const unsigned*)(p + 2 * v + (v >= 4 ? 8 : 0));
}

__device__ __forceinline__ void load16f(const bf16_t* __restrict__ p, float* f) {
  const unsigned* up = (const unsigned*)p;
#pragma unroll
  for (int i = 0; i < 8; ++i) {
    BPack pk; pk.u = up[i];
    f[2*i]   = (float)pk.h[0];
    f[2*i+1] = (float)pk.h[1];
  }
}

__device__ __forceinline__ unsigned pack2(float a, float b) {
  BPack pk; pk.h[0] = (bf16_t)a; pk.h[1] = (bf16_t)b; return pk.u;
}

// ------------------------------ small helpers ------------------------------

__global__ __launch_bounds__(256) void cvt_kernel(const float* __restrict__ in,
                                                  bf16_t* __restrict__ out, int n) {
  int i = blockIdx.x * 256 + threadIdx.x;
  if (i < n) out[i] = (bf16_t)in[i];
}

__global__ void zero_kernel(unsigned* __restrict__ p, int n) {
  int i = blockIdx.x * 64 + threadIdx.x;
  if (i < n) p[i] = 0u;
}

// conv3_w (96,32,3,3,3) [oc][ic][tap]  ->  Wt[oc][tap*32 + ic]  (tap-major K)
__global__ __launch_bounds__(256) void conv3_tr_kernel(const float* __restrict__ in,
                                                       bf16_t* __restrict__ out) {
  int i = blockIdx.x * 256 + threadIdx.x;
  if (i >= 96 * 864) return;
  int oc = i / 864, r = i % 864;
  int tap = r >> 5, ic = r & 31;
  out[i] = (bf16_t)in[oc * 864 + ic * 27 + tap];
}

// LayerNorm over C=96, one wave32 per token, output bf16.
__global__ __launch_bounds__(256) void ln_kernel(const float* __restrict__ x,
                                                 const float* __restrict__ w,
                                                 const float* __restrict__ b,
                                                 bf16_t* __restrict__ out) {
  int tok  = (blockIdx.x * 256 + threadIdx.x) >> 5;
  int lane = threadIdx.x & 31;
  if (tok >= M_TOK) return;
  const float* xp = x + (size_t)tok * 96;
  float v0 = xp[lane], v1 = xp[lane + 32], v2 = xp[lane + 64];
  float s = v0 + v1 + v2, s2 = v0*v0 + v1*v1 + v2*v2;
#pragma unroll
  for (int m = 16; m > 0; m >>= 1) { s += __shfl_xor(s, m); s2 += __shfl_xor(s2, m); }
  float mu = s * (1.0f / 96.0f);
  float rs = rsqrtf(s2 * (1.0f / 96.0f) - mu * mu + 1e-5f);
  bf16_t* op = out + (size_t)tok * 96;
  op[lane]      = (bf16_t)((v0 - mu) * rs * w[lane]      + b[lane]);
  op[lane + 32] = (bf16_t)((v1 - mu) * rs * w[lane + 32] + b[lane + 32]);
  op[lane + 64] = (bf16_t)((v2 - mu) * rs * w[lane + 64] + b[lane + 64]);
}

// ------------------------------ generic GEMM -------------------------------
// C[M][N] = A_bf16[M][K] @ Bt_bf16[N][K]^T, one 16x32 tile per wave.
template<bool BIAS, bool GELU, bool RES, bool OUT32, bool OUT16>
__global__ __launch_bounds__(256)
void gemm_bf16_kernel(const bf16_t* __restrict__ A, const bf16_t* __restrict__ Bt,
                      const float* __restrict__ bias, const float* __restrict__ res,
                      float* __restrict__ out32, bf16_t* __restrict__ out16,
                      int M, int N, int K) {
  int lane = threadIdx.x & 31;
  int gw   = (blockIdx.x * 256 + threadIdx.x) >> 5;
  int colTiles = N >> 5;
  int mTile = gw / colTiles;
  int nTile = gw % colTiles;
  if (mTile * 16 >= M) return;            // wave-uniform: EXEC stays all-ones
  int m0 = mTile * 16, n0 = nTile * 32;

  f32x8 acc0 = {0.f,0.f,0.f,0.f,0.f,0.f,0.f,0.f};
  f32x8 acc1 = {0.f,0.f,0.f,0.f,0.f,0.f,0.f,0.f};
  for (int k0 = 0; k0 < K; k0 += 32) {
    Frag a, b0, b1;
    load_frag(a,  A,  m0,      K, k0, lane);
    load_frag(b0, Bt, n0,      K, k0, lane);
    load_frag(b1, Bt, n0 + 16, K, k0, lane);
    acc0 = __builtin_amdgcn_wmma_f32_16x16x32_bf16(false, a.v, false, b0.v,
                                                   (short)0, acc0, false, false);
    acc1 = __builtin_amdgcn_wmma_f32_16x16x32_bf16(false, a.v, false, b1.v,
                                                   (short)0, acc1, false, false);
  }
  // D layout: VGPR r -> M = m0 + r + 8*(lane>=16), N = n0(+16) + (lane&15)
  int mrow = m0 + ((lane >> 4) << 3);
  int ncol = lane & 15;
#pragma unroll
  for (int r = 0; r < 8; ++r) {
    int m = mrow + r;
#pragma unroll
    for (int t = 0; t < 2; ++t) {
      int n = n0 + t * 16 + ncol;
      float v = (t == 0) ? acc0[r] : acc1[r];
      if (BIAS) v += bias[n];
      if (GELU) v = 0.5f * v * (1.0f + erff(v * 0.70710678118654752f));
      size_t idx = (size_t)m * N + n;
      if (RES)   v += res[idx];
      if (OUT32) out32[idx] = v;
      if (OUT16) out16[idx] = (bf16_t)v;
    }
  }
}

// --------------------------- dilate attention ------------------------------
// One thread per (window, n, dilation, head).  Softmax has 9 slots; the 6
// empty slots contribute exp(0)=1 to the denominator, padded taps contribute
// exp(0)=1 with v=0.
__global__ __launch_bounds__(256)
void attn_kernel(const bf16_t* __restrict__ qkv, bf16_t* __restrict__ out) {
  int idx = blockIdx.x * 256 + threadIdx.x;       // grid is exact: 2048*98*6
  int t6 = idx % 6;
  int n  = (idx / 6) % 98;
  int w  = idx / (6 * 98);
  int di = t6 >> 1, hh = t6 & 1;
  int dil = di + 1;                               // DILATIONS = (1,2,3)
  int b = w >> 9, r = w & 511;                    // windows: (B,8,8,8)
  int wd = r >> 6, wh = (r >> 3) & 7, ww = r & 7;
  int ch = di * 32 + hh * 16;

  // token index of window-local position n
  int td = n / 49, r2 = n % 49;
  int d0 = wd * 2 + td, h0 = wh * 7 + r2 / 7, x0 = ww * 7 + r2 % 7;
  size_t tok = ((size_t)(b * 16 + d0) * 56 + h0) * 56 + x0;

  float q[16];
  load16f(qkv + tok * 288 + ch, q);

  float vout[16];
#pragma unroll
  for (int c = 0; c < 16; ++c) vout[c] = 0.0f;
  float esum = 6.0f;                              // 6 always-zero slots
#pragma unroll
  for (int j = 0; j < 3; ++j) {
    int nn = n + (j - 1) * dil;
    if (nn >= 0 && nn < 98) {
      int tdn = nn / 49, rn = nn % 49;
      int dn = wd * 2 + tdn, hn = wh * 7 + rn / 7, xn = ww * 7 + rn % 7;
      size_t tb = (((size_t)(b * 16 + dn) * 56 + hn) * 56 + xn) * 288;
      float kv[16], vv[16];
      load16f(qkv + tb + 96  + ch, kv);
      load16f(qkv + tb + 192 + ch, vv);
      float dot = 0.0f;
#pragma unroll
      for (int c = 0; c < 16; ++c) dot += q[c] * kv[c];
      float ej = __expf(dot * 0.25f);             // scale = head_dim^-0.5
#pragma unroll
      for (int c = 0; c < 16; ++c) vout[c] += ej * vv[c];
      esum += ej;
    } else {
      esum += 1.0f;                               // zero-padded tap: exp(0)
    }
  }
  float inv = 1.0f / esum;
  unsigned* op = (unsigned*)(out + tok * 96 + ch);
#pragma unroll
  for (int i = 0; i < 8; ++i)
    op[i] = pack2(vout[2*i] * inv, vout[2*i+1] * inv);
}

// -------------------- grouped 3x3x3 conv as implicit GEMM ------------------
// Per wave: 16 tokens x 32 out-channels of one group; K = 27 taps x 32 ic.
// 'SAME' zero padding is implemented by per-lane address selection against a
// 256-byte zero page: loads are unconditional (no EXEC divergence, coalesced
// into b128 pairs), out-of-bounds taps simply read zeros.
__global__ __launch_bounds__(256)
void conv3_kernel(const bf16_t* __restrict__ x1, const bf16_t* __restrict__ Wt,
                  const bf16_t* __restrict__ zpage, bf16_t* __restrict__ out) {
  int lane = threadIdx.x & 31;
  int gw   = (blockIdx.x * 256 + threadIdx.x) >> 5;
  int mTile = gw / 3;
  int g     = gw % 3;
  int m0 = mTile * 16;
  if (m0 >= M_TOK) return;

  int m = m0 + (lane & 15);                       // this lane's token row
  int t = m;
  int wI = t % 56; t /= 56;
  int hI = t % 56; t /= 56;
  int dI = t % 16; int bI = t / 16;
  int kb8 = (lane >> 4) << 3;                     // +8 K offset for hi lanes
  // pointer to this lane's center-tap 32-channel slice
  const bf16_t* center = x1 + (size_t)m * 96 + g * 32 + kb8;

  f32x8 acc0 = {0.f,0.f,0.f,0.f,0.f,0.f,0.f,0.f};
  f32x8 acc1 = {0.f,0.f,0.f,0.f,0.f,0.f,0.f,0.f};
#pragma unroll
  for (int tap = 0; tap < 27; ++tap) {
    const int kd = tap / 9, kh = (tap / 3) % 3, kw = tap % 3;
    int dd = dI + kd - 1, hh = hI + kh - 1, wv = wI + kw - 1;
    bool ok = ((unsigned)dd < 16u) & ((unsigned)hh < 56u) & ((unsigned)wv < 56u);
    const int delta = ((kd - 1) * 3136 + (kh - 1) * 56 + (kw - 1)) * 96;
    const bf16_t* ap = ok ? (center + delta) : zpage;   // address select
    Frag a, b0, b1;
#pragma unroll
    for (int v = 0; v < 8; ++v)
      a.u[v] = *(const unsigned*)(ap + 2 * v + (v >= 4 ? 8 : 0));
    const int k0 = tap * 32;
    load_frag(b0, Wt, g * 32,      864, k0, lane);
    load_frag(b1, Wt, g * 32 + 16, 864, k0, lane);
    acc0 = __builtin_amdgcn_wmma_f32_16x16x32_bf16(false, a.v, false, b0.v,
                                                   (short)0, acc0, false, false);
    acc1 = __builtin_amdgcn_wmma_f32_16x16x32_bf16(false, a.v, false, b1.v,
                                                   (short)0, acc1, false, false);
  }
  int mrow = m0 + ((lane >> 4) << 3);
  int ncol = lane & 15;
#pragma unroll
  for (int r = 0; r < 8; ++r) {
#pragma unroll
    for (int tt = 0; tt < 2; ++tt) {
      int chn = g * 32 + tt * 16 + ncol;
      float v = (tt == 0) ? acc0[r] : acc1[r];
      out[(size_t)(mrow + r) * 96 + chn] = (bf16_t)v;
    }
  }
}

// ------------------------------ GroupNorm ----------------------------------
// Deterministic two-stage reduction over (24 channels x 50176 tokens) per
// (batch, group): 120 partial blocks each, then a tiny reduce kernel.
#define GN_BLK 120

__global__ __launch_bounds__(256)
void gn_stats_kernel(const bf16_t* __restrict__ conv, float* __restrict__ part) {
  __shared__ float ssum[8], ssum2[8];
  int bg  = blockIdx.x / GN_BLK;
  int blk = blockIdx.x % GN_BLK;
  int b = bg >> 2, g = bg & 3;
  float s = 0.0f, s2 = 0.0f;
  for (int tok = blk * 256 + threadIdx.x; tok < NTOK_B; tok += GN_BLK * 256) {
    const bf16_t* p = conv + (size_t)(b * NTOK_B + tok) * 96 + g * 24;
#pragma unroll
    for (int c = 0; c < 24; c += 2) {               // 12 contiguous dwords
      BPack pk; pk.u = *(const unsigned*)(p + c);
      float a0 = (float)pk.h[0], a1 = (float)pk.h[1];
      s += a0 + a1; s2 += a0 * a0 + a1 * a1;
    }
  }
#pragma unroll
  for (int m = 16; m > 0; m >>= 1) { s += __shfl_xor(s, m); s2 += __shfl_xor(s2, m); }
  int wid = threadIdx.x >> 5;
  if ((threadIdx.x & 31) == 0) { ssum[wid] = s; ssum2[wid] = s2; }
  __syncthreads();
  if (threadIdx.x == 0) {
    float ts = 0.0f, ts2 = 0.0f;
#pragma unroll
    for (int i = 0; i < 8; ++i) { ts += ssum[i]; ts2 += ssum2[i]; }
    part[(bg * GN_BLK + blk) * 2]     = ts;
    part[(bg * GN_BLK + blk) * 2 + 1] = ts2;
  }
}

__global__ void gn_reduce_kernel(const float* __restrict__ part,
                                 float* __restrict__ stats) {
  int bg = threadIdx.x;
  if (bg >= 16) return;
  float s = 0.0f, s2 = 0.0f;
  for (int i = 0; i < GN_BLK; ++i) {
    s  += part[(bg * GN_BLK + i) * 2];
    s2 += part[(bg * GN_BLK + i) * 2 + 1];
  }
  stats[2 * bg] = s; stats[2 * bg + 1] = s2;
}

__global__ __launch_bounds__(256)
void gn_apply_kernel(const bf16_t* __restrict__ conv, const float* __restrict__ stats,
                     const float* __restrict__ gw, const float* __restrict__ gb,
                     bf16_t* __restrict__ out) {
  int idx = blockIdx.x * 256 + threadIdx.x;       // grid exact: M_TOK*96/256
  int chn = idx % 96;
  int tok = idx / 96;
  int b = tok / NTOK_B, g = chn / 24;
  int bg = b * 4 + g;
  const float invN = 1.0f / (float)(NTOK_B * 24);
  float mu  = stats[2 * bg] * invN;
  float var = stats[2 * bg + 1] * invN - mu * mu;
  float v = (float)conv[idx];
  float y = (v - mu) * rsqrtf(var + 1e-5f) * gw[chn] + gb[chn];
  out[idx] = (bf16_t)fmaxf(y, 0.0f);              // ReLU
}

// ------------------------------- launcher ----------------------------------

extern "C" void kernel_launch(void* const* d_in, const int* in_sizes, int n_in,
                              void* d_out, int out_size, void* d_ws, size_t ws_size,
                              hipStream_t stream) {
  const float* x       = (const float*)d_in[0];
  const float* n1w     = (const float*)d_in[1];
  const float* n1b     = (const float*)d_in[2];
  const float* qkv_w   = (const float*)d_in[3];
  const float* proj_w  = (const float*)d_in[4];
  const float* proj_b  = (const float*)d_in[5];
  const float* conv3_w = (const float*)d_in[6];
  const float* gn_w    = (const float*)d_in[7];
  const float* gn_b    = (const float*)d_in[8];
  const float* convp_w = (const float*)d_in[9];
  const float* n2w     = (const float*)d_in[10];
  const float* n2b     = (const float*)d_in[11];
  const float* fc1_w   = (const float*)d_in[12];
  const float* fc1_b   = (const float*)d_in[13];
  const float* fc2_w   = (const float*)d_in[14];
  const float* fc2_b   = (const float*)d_in[15];
  float* out = (float*)d_out;

  char* ws = (char*)d_ws;
  size_t off = 0;
  auto alloc = [&](size_t bytes) {
    void* p = ws + off;
    off += (bytes + 255) & ~(size_t)255;
    return p;
  };
  // weights (bf16) + zero page + GN scratch
  bf16_t* qkv_wt   = (bf16_t*)alloc(288 * 96 * 2);
  bf16_t* proj_wt  = (bf16_t*)alloc(96 * 96 * 2);
  bf16_t* fc1_wt   = (bf16_t*)alloc(384 * 96 * 2);
  bf16_t* fc2_wt   = (bf16_t*)alloc(96 * 384 * 2);
  bf16_t* convp_wt = (bf16_t*)alloc(96 * 96 * 2);
  bf16_t* conv3_wt = (bf16_t*)alloc(96 * 864 * 2);
  bf16_t* zpage    = (bf16_t*)alloc(256);
  float*  gn_part  = (float*)alloc(16 * GN_BLK * 2 * 4);
  float*  gn_stats = (float*)alloc(32 * 4);
  // activation buffers (lifetimes disjoint where shared)
  bf16_t* R1 = (bf16_t*)alloc((size_t)M_TOK * 96 * 2);   // xn1 / relu / xn2
  bf16_t* R2 = (bf16_t*)alloc((size_t)M_TOK * 384 * 2);  // qkv (288) / mlp hidden (384)
  bf16_t* R3 = (bf16_t*)alloc((size_t)M_TOK * 96 * 2);   // attention output
  bf16_t* R4 = (bf16_t*)alloc((size_t)M_TOK * 96 * 2);   // x1 bf16 (conv input)
  bf16_t* R5 = (bf16_t*)alloc((size_t)M_TOK * 96 * 2);   // grouped-conv output

  // 0) weights -> bf16 (conv3 also transposed to tap-major K); zero the page
  cvt_kernel<<<(288*96 + 255)/256, 256, 0, stream>>>(qkv_w,   qkv_wt,   288*96);
  cvt_kernel<<<(96*96  + 255)/256, 256, 0, stream>>>(proj_w,  proj_wt,  96*96);
  cvt_kernel<<<(384*96 + 255)/256, 256, 0, stream>>>(fc1_w,   fc1_wt,   384*96);
  cvt_kernel<<<(96*384 + 255)/256, 256, 0, stream>>>(fc2_w,   fc2_wt,   96*384);
  cvt_kernel<<<(96*96  + 255)/256, 256, 0, stream>>>(convp_w, convp_wt, 96*96);
  conv3_tr_kernel<<<(96*864 + 255)/256, 256, 0, stream>>>(conv3_w, conv3_wt);
  zero_kernel<<<1, 64, 0, stream>>>((unsigned*)zpage, 64);

  const int ln_grid = M_TOK / 8;            // 1 wave per token, 8 waves/block
  const int mt      = M_TOK / 16;           // 12544 row tiles

  // 1) LN1 -> bf16
  ln_kernel<<<ln_grid, 256, 0, stream>>>(x, n1w, n1b, R1);

  // 2) qkv GEMM: [M,96] @ [96,288]
  gemm_bf16_kernel<false,false,false,false,true>
      <<<mt * (288/32) / 8, 256, 0, stream>>>(R1, qkv_wt, nullptr, nullptr,
                                              nullptr, R2, M_TOK, 288, 96);

  // 3) multi-dilate local attention (3 dilations x 2 heads of dim 16)
  attn_kernel<<<2048 * 98 * 6 / 256, 256, 0, stream>>>(R2, R3);

  // 4) proj GEMM + bias + residual(x): fp32 -> d_out, bf16 copy -> R4
  gemm_bf16_kernel<true,false,true,true,true>
      <<<mt * (96/32) / 8, 256, 0, stream>>>(R3, proj_wt, proj_b, x,
                                             out, R4, M_TOK, 96, 96);

  // 5) grouped 3x3x3 conv (implicit GEMM, groups=3) -> R5
  conv3_kernel<<<mt * 3 / 8, 256, 0, stream>>>(R4, conv3_wt, zpage, R5);

  // 6) GroupNorm(4) stats (deterministic two-stage), apply + ReLU -> R1
  gn_stats_kernel<<<16 * GN_BLK, 256, 0, stream>>>(R5, gn_part);
  gn_reduce_kernel<<<1, 32, 0, stream>>>(gn_part, gn_stats);
  gn_apply_kernel<<<(size_t)M_TOK * 96 / 256, 256, 0, stream>>>(R5, gn_stats,
                                                                gn_w, gn_b, R1);

  // 7) 1x1 conv GEMM + residual(d_out) -> d_out
  gemm_bf16_kernel<false,false,true,true,false>
      <<<mt * (96/32) / 8, 256, 0, stream>>>(R1, convp_wt, nullptr, out,
                                             out, nullptr, M_TOK, 96, 96);

  // 8) LN2 -> bf16 (R1 free again)
  ln_kernel<<<ln_grid, 256, 0, stream>>>(out, n2w, n2b, R1);

  // 9) fc1 GEMM + bias + exact GELU -> R2 (bf16 hidden)
  gemm_bf16_kernel<true,true,false,false,true>
      <<<mt * (384/32) / 8, 256, 0, stream>>>(R1, fc1_wt, fc1_b, nullptr,
                                              nullptr, R2, M_TOK, 384, 96);

  // 10) fc2 GEMM + bias + residual(d_out) -> d_out
  gemm_bf16_kernel<true,false,true,true,false>
      <<<mt * (96/32) / 8, 256, 0, stream>>>(R2, fc2_wt, fc2_b, out,
                                             out, nullptr, M_TOK, 96, 384);
}
